// MNNDiffusionFilter_42356967473546
// MI455X (gfx1250) — matrix operand order repaired and checked by reading, not compiled, not run on hardware
//
#include <hip/hip_runtime.h>

// CDNA5 / gfx1250: wave32, WMMA bf16 16x16x32, f32 accumulate.
// Pk streamed once from HBM (1 GiB) -> memory-bound target ~46us @ 23.3 TB/s.

typedef __attribute__((ext_vector_type(16))) __bf16 v16bf;
typedef __attribute__((ext_vector_type(8)))  __bf16 v8bf;
typedef __attribute__((ext_vector_type(8)))  float  v8f;

#define N_DIM 4096
#define F_DIM 64
#define K_DIM 16
#define C_DIM 16
#define MT    128     // m rows staged to LDS per round (4 WMMA k-steps of 32)
#define XT_PITCH 136  // bf16 per xT row: 128 + 8 pad -> 272B rows (16B aligned)
#define DF_PITCH 20   // floats per (k,n_local) row in epilogue tile: 16 + pad

__global__ __launch_bounds__(256)
void mnn_diffusion_wmma(const float* __restrict__ Pk,
                        const float* __restrict__ x,
                        const float* __restrict__ alpha,
                        float* __restrict__ out)
{
    __shared__ __bf16 xT[F_DIM * XT_PITCH];         // 17408 B : x chunk, transposed, bf16
    __shared__ float  dsm[K_DIM * 16 * DF_PITCH];   // 20480 B : diff tile for one f-tile
    __shared__ float  s_alpha[C_DIM * K_DIM];       // 1024 B

    const int tid  = threadIdx.x;
    const int lane = tid & 31;
    const int col  = lane & 15;   // N-column (B/D) or M-row (A) inside tile
    const int hi   = lane >> 4;   // lane half selects K/M sub-range per ISA layout
    const int wv   = tid >> 5;    // wave 0..7
    const int n0   = blockIdx.x << 4;

    s_alpha[tid] = alpha[tid];    // C*K == 256 == blockDim

    const int k0 = wv;            // this wave's two diffusion hops
    const int k1 = wv + 8;
    const int row = n0 + col;
    const float* aP0 = Pk + ((size_t)k0 * N_DIM + row) * N_DIM + (hi << 3);
    const float* aP1 = Pk + ((size_t)k1 * N_DIM + row) * N_DIM + (hi << 3);

    const v8f vz = {0.f,0.f,0.f,0.f,0.f,0.f,0.f,0.f};
    v8f acc0[4], acc1[4];
    #pragma unroll
    for (int ft = 0; ft < 4; ++ft) { acc0[ft] = vz; acc1[ft] = vz; }

    // x staging: thread covers f columns xf..xf+3 for rows xr, xr+16, ... xr+112
    const int xr = tid >> 4;          // 0..15
    const int xf = (tid & 15) << 2;   // 0..60

    for (int m = 0; m < N_DIM; m += MT) {
        __syncthreads();  // previous round's xT readers done
        #pragma unroll
        for (int rr = 0; rr < MT; rr += 16) {
            const int r = rr + xr;
            const float4 v0 = *(const float4*)(x + (size_t)(m + r) * F_DIM + xf);
            xT[(xf + 0) * XT_PITCH + r] = (__bf16)v0.x;
            xT[(xf + 1) * XT_PITCH + r] = (__bf16)v0.y;
            xT[(xf + 2) * XT_PITCH + r] = (__bf16)v0.z;
            xT[(xf + 3) * XT_PITCH + r] = (__bf16)v0.w;
        }
        __syncthreads();

        #pragma unroll
        for (int mm = 0; mm < MT; mm += 32) {
            // A tiles (Pk rows, bf16). ISA 16-bit A 16x32 layout:
            // lanes 0-15: elem 0..7 -> K=0..7,  elem 8..15 -> K=16..23
            // lanes 16-31: elem 0..7 -> K=8..15, elem 8..15 -> K=24..31
            v16bf a0, a1;
            {
                const float4 q0 = *(const float4*)(aP0 + m + mm);
                const float4 q1 = *(const float4*)(aP0 + m + mm + 4);
                const float4 q2 = *(const float4*)(aP0 + m + mm + 16);
                const float4 q3 = *(const float4*)(aP0 + m + mm + 20);
                a0[0]=(__bf16)q0.x;  a0[1]=(__bf16)q0.y;  a0[2]=(__bf16)q0.z;  a0[3]=(__bf16)q0.w;
                a0[4]=(__bf16)q1.x;  a0[5]=(__bf16)q1.y;  a0[6]=(__bf16)q1.z;  a0[7]=(__bf16)q1.w;
                a0[8]=(__bf16)q2.x;  a0[9]=(__bf16)q2.y;  a0[10]=(__bf16)q2.z; a0[11]=(__bf16)q2.w;
                a0[12]=(__bf16)q3.x; a0[13]=(__bf16)q3.y; a0[14]=(__bf16)q3.z; a0[15]=(__bf16)q3.w;
            }
            {
                const float4 q0 = *(const float4*)(aP1 + m + mm);
                const float4 q1 = *(const float4*)(aP1 + m + mm + 4);
                const float4 q2 = *(const float4*)(aP1 + m + mm + 16);
                const float4 q3 = *(const float4*)(aP1 + m + mm + 20);
                a1[0]=(__bf16)q0.x;  a1[1]=(__bf16)q0.y;  a1[2]=(__bf16)q0.z;  a1[3]=(__bf16)q0.w;
                a1[4]=(__bf16)q1.x;  a1[5]=(__bf16)q1.y;  a1[6]=(__bf16)q1.z;  a1[7]=(__bf16)q1.w;
                a1[8]=(__bf16)q2.x;  a1[9]=(__bf16)q2.y;  a1[10]=(__bf16)q2.z; a1[11]=(__bf16)q2.w;
                a1[12]=(__bf16)q3.x; a1[13]=(__bf16)q3.y; a1[14]=(__bf16)q3.z; a1[15]=(__bf16)q3.w;
            }

            // B tiles (x, bf16). ISA 16-bit B 32x16 layout: lane = column,
            // same K order as A: elem 0..7 -> K=8*hi+0..7, elem 8..15 -> K=16+8*hi+0..7.
            #pragma unroll
            for (int ft = 0; ft < 4; ++ft) {
                const __bf16* bp = &xT[((ft << 4) + col) * XT_PITCH + mm + (hi << 3)];
                const v8bf b_lo = *(const v8bf*)bp;        // ds_load_b128
                const v8bf b_hi = *(const v8bf*)(bp + 16); // ds_load_b128
                v16bf b;
                #pragma unroll
                for (int i = 0; i < 8; ++i) { b[i] = b_lo[i]; b[i + 8] = b_hi[i]; }
                acc0[ft] = __builtin_amdgcn_wmma_f32_16x16x32_bf16(
                    false, a0, false, b, (short)0, acc0[ft], false, false);
                acc1[ft] = __builtin_amdgcn_wmma_f32_16x16x32_bf16(
                    false, a1, false, b, (short)0, acc1[ft], false, false);
            }
        }
    }

    // Epilogue (fully unrolled -> constant register indices, no v_movrels):
    // per f-tile, gather diff[k, n_local, f_local] across waves via LDS, then
    // out[n, f, c] = sum_k alpha[c,k] * diff[k,n,f].
    const int nl = tid >> 4;   // n_local 0..15
    const int fl = tid & 15;   // f_local 0..15
    #pragma unroll
    for (int ft = 0; ft < 4; ++ft) {
        __syncthreads();
        // D layout: VGPR j -> M = j + 8*hi, N = col
        #pragma unroll
        for (int j = 0; j < 8; ++j) {
            dsm[((k0 << 4) + (hi << 3) + j) * DF_PITCH + col] = acc0[ft][j];
            dsm[((k1 << 4) + (hi << 3) + j) * DF_PITCH + col] = acc1[ft][j];
        }
        __syncthreads();

        float d[16];
        #pragma unroll
        for (int k = 0; k < 16; ++k) d[k] = dsm[((k << 4) + nl) * DF_PITCH + fl];
        float o[16];
        #pragma unroll
        for (int c = 0; c < 16; ++c) {
            float s = 0.f;
            #pragma unroll
            for (int k = 0; k < 16; ++k) s = fmaf(d[k], s_alpha[(c << 4) + k], s);
            o[c] = s;
        }
        float* op = out + ((size_t)(n0 + nl) * F_DIM + (ft << 4) + fl) * C_DIM;
        #pragma unroll
        for (int c = 0; c < 16; c += 4)
            *(float4*)(op + c) = make_float4(o[c], o[c + 1], o[c + 2], o[c + 3]);
    }
}

extern "C" void kernel_launch(void* const* d_in, const int* in_sizes, int n_in,
                              void* d_out, int out_size, void* d_ws, size_t ws_size,
                              hipStream_t stream) {
    (void)in_sizes; (void)n_in; (void)out_size; (void)d_ws; (void)ws_size;
    const float* Pk    = (const float*)d_in[0];
    const float* x     = (const float*)d_in[1];
    const float* alpha = (const float*)d_in[2];
    float* out = (float*)d_out;

    dim3 grid(N_DIM / 16);   // 256 n-tiles
    dim3 block(256);         // 8 wave32 waves: wave w owns hops k=w, k=w+8
    hipLaunchKernelGGL(mnn_diffusion_wmma, grid, block, 0, stream, Pk, x, alpha, out);
}
